// PairwiseGeometricBilinear_24592982737128
// MI455X (gfx1250) — matrix operand order repaired and checked by reading, not compiled, not run on hardware
//
#include <hip/hip_runtime.h>
#include <stdint.h>

// ---------------------------------------------------------------------------
// PGA(3,0,1) blade tables, generated at compile time (mirrors the reference).
// ---------------------------------------------------------------------------
constexpr int MASKS[16]  = {0,1,2,4,8,3,5,9,6,10,12,7,11,13,14,15};
constexpr int IDXOF[16]  = {0,1,2,5,3,6,8,11,4,7,9,12,10,13,14,15}; // mask -> index
constexpr int GRADE[16]  = {0,1,1,1,1,2,2,2,2,2,2,3,3,3,3,4};
constexpr int HAS_E0[16] = {0,1,0,0,0,1,1,1,0,0,0,1,1,1,0,1};
constexpr int SRC_K[16]  = {0,0,2,3,4,2,3,4,8,9,10,8,9,10,14,14};
constexpr int E0W[16]    = {0,5,0,0,0,6,6,6,0,0,0,7,7,7,0,8};

constexpr float reorder_sign(int a, int b) {
  int s = 0; a >>= 1;
  while (a) { s += __builtin_popcount(a & b); a >>= 1; }
  return (s & 1) ? -1.0f : 1.0f;
}

struct BilTab { float s[16][16]; int j[16][16]; };

constexpr BilTab make_gp() {
  BilTab t{};
  for (int k = 0; k < 16; ++k)
    for (int i = 0; i < 16; ++i) {
      int mi = MASKS[i], mk = MASKS[k];
      int mj = mi ^ mk;
      t.j[k][i] = IDXOF[mj];
      t.s[k][i] = ((mi & mj & 1) == 0) ? reorder_sign(mi, mj) : 0.0f;
    }
  return t;
}

constexpr BilTab make_join() {
  BilTab t{};
  for (int k = 0; k < 16; ++k)
    for (int i = 0; i < 16; ++i) {
      int mi = MASKS[i], mk = MASKS[k];
      int mj = mi ^ mk ^ 15;            // forced by k = dual(idx(mi^mj))
      t.j[k][i] = IDXOF[mj];
      if ((mi | mj) == 15) {            // dual masks disjoint
        int da = mi ^ 15, db = mj ^ 15;
        int mm = da ^ db;               // == mi ^ mj
        t.s[k][i] = reorder_sign(mi, da)       // D[dual(i), i]
                  * reorder_sign(mj, db)       // D[dual(j), j]
                  * reorder_sign(da, db)       // wedge[da, db, m]
                  * reorder_sign(mm, mm ^ 15); // D[k, m]
      } else {
        t.s[k][i] = 0.0f;
      }
    }
  return t;
}

constexpr BilTab GP_T   = make_gp();
constexpr BilTab JOIN_T = make_join();

// ---------------------------------------------------------------------------
// WMMA plumbing (CDNA5: V_WMMA_F32_16X16X32_BF16, wave32)
// ---------------------------------------------------------------------------
typedef __attribute__((ext_vector_type(16))) __bf16 v16bf;
typedef __attribute__((ext_vector_type(8)))  float  v8f;

__device__ inline v8f wmma_bf16(v16bf a, v16bf b, v8f c) {
  return __builtin_amdgcn_wmma_f32_16x16x32_bf16(
      /*neg_a=*/false, a, /*neg_b=*/false, b,
      /*c_mod=*/(short)0, c, /*reuse_a=*/false, /*reuse_b=*/false);
}

#define TILE  16
#define MROW  1044   // padded floats per position row in LDS (bank-conflict-free, 16B aligned)

// A-fragment (16x32 bf16): lane ln: M = ln&15, K-half = ln>>4.
// element t maps to K = (t&7) + 8*half + 16*(t>>3) within the 32-wide K block.
template <typename T>
__device__ inline v16bf load_a_frag(const T* buf, int comp, int kb, int m16, int hf) {
  v16bf a;
  const T* base = buf + m16 * MROW + comp;
#pragma unroll
  for (int t = 0; t < 16; ++t) {
    int K = (t & 7) + 8 * hf + ((t >> 3) << 4) + kb * 32;
    a[t] = (__bf16)(float)base[K * 16];
  }
  return a;
}

// B-fragments are pre-packed per (tensor, grade, kb, nb): 32 lanes x 16 bf16, contiguous.
__device__ inline v16bf load_b_frag(const __bf16* wp, int wsel, int g, int kb, int nb, int ln) {
  const __bf16* p = wp + ((((size_t)(wsel * 9 + g) * 2 + kb) * 4 + nb) * 32 + ln) * 16;
  return *(const v16bf*)p;   // 32B aligned
}

// D (16x16 f32): VGPR r -> M = r + 8*(ln>>4), N = ln&15
__device__ inline void store_d(float* buf, int comp, int nb, int m16, int hf, v8f acc) {
#pragma unroll
  for (int r = 0; r < 8; ++r) {
    int M = r + 8 * hf;
    int o = nb * 16 + m16;
    buf[M * MROW + o * 16 + comp] = acc[r];
  }
}

// One equi_linear component: D_k = X_k * W[g(k)]^T (+ X_src(k) * W[e0w(k)]^T) (+ bias if k==0)
template <typename T>
__device__ inline void equi_comp(const T* xbuf, const __bf16* wp, int wsel,
                                 const float* bias, int k,
                                 int m16, int hf, int ln, v8f acc[4]) {
  v16bf xa0 = load_a_frag(xbuf, k, 0, m16, hf);
  v16bf xa1 = load_a_frag(xbuf, k, 1, m16, hf);
  const int e0 = HAS_E0[k];
  v16bf xs0 = xa0, xs1 = xa1;
  if (e0) {
    xs0 = load_a_frag(xbuf, SRC_K[k], 0, m16, hf);
    xs1 = load_a_frag(xbuf, SRC_K[k], 1, m16, hf);
  }
  const int g = GRADE[k], ge = E0W[k];
#pragma unroll
  for (int nb = 0; nb < 4; ++nb) {
    v8f a = {};
    a = wmma_bf16(xa0, load_b_frag(wp, wsel, g, 0, nb, ln), a);
    a = wmma_bf16(xa1, load_b_frag(wp, wsel, g, 1, nb, ln), a);
    if (e0) {
      a = wmma_bf16(xs0, load_b_frag(wp, wsel, ge, 0, nb, ln), a);
      a = wmma_bf16(xs1, load_b_frag(wp, wsel, ge, 1, nb, ln), a);
    }
    if (k == 0) {
      float bv = bias[nb * 16 + m16];
#pragma unroll
      for (int rr = 0; rr < 8; ++rr) a[rr] += bv;
    }
    acc[nb] = a;
  }
}

template <bool USE_GP>
__device__ inline void bil_eval(const float* l, const float* r, float* o) {
#pragma unroll
  for (int k = 0; k < 16; ++k) {
    float acc = 0.0f;
#pragma unroll
    for (int i = 0; i < 16; ++i) {
      const float s = USE_GP ? GP_T.s[k][i] : JOIN_T.s[k][i];
      const int   j = USE_GP ? GP_T.j[k][i] : JOIN_T.j[k][i];
      if (s != 0.0f) acc += s * l[i] * r[j];
    }
    o[k] = acc;
  }
}

// ---------------------------------------------------------------------------
// Async bulk copies global <-> LDS (CDNA5 ASYNCcnt path, no VGPR round-trip).
// GVS mode: uniform SGPR 64-bit base + per-lane 32-bit byte offset.
// ---------------------------------------------------------------------------
__device__ inline void async_wait_all() {
  asm volatile("s_wait_asynccnt 0x0" ::: "memory");
}

__device__ inline void async_tile_load(float* lds_buf, const float* gbase, int tid) {
#pragma unroll
  for (int it = 0; it < 16; ++it) {
    int idx = tid + it * 256;
    int m = idx >> 8, j = idx & 255;
    uint32_t lds_off = (uint32_t)(uintptr_t)(lds_buf + m * MROW + j * 4);
    uint32_t goff    = (uint32_t)((m * 1024 + j * 4) * 4);
    asm volatile("global_load_async_to_lds_b128 %0, %1, %2"
                 :
                 : "v"(lds_off), "v"(goff), "s"(gbase)
                 : "memory");
  }
  async_wait_all();
}

__device__ inline void async_tile_store(float* gbase, const float* lds_buf, int tid) {
#pragma unroll
  for (int it = 0; it < 16; ++it) {
    int idx = tid + it * 256;
    int m = idx >> 8, j = idx & 255;
    uint32_t lds_off = (uint32_t)(uintptr_t)(lds_buf + m * MROW + j * 4);
    uint32_t goff    = (uint32_t)((m * 1024 + j * 4) * 4);
    asm volatile("global_store_async_from_lds_b128 %0, %1, %2"
                 :
                 : "v"(goff), "v"(lds_off), "s"(gbase)
                 : "memory");
  }
  async_wait_all();
}

// ---------------------------------------------------------------------------
// Weight repack: f32 (C,C,9) -> bf16 WMMA-B fragments for 3 tensors x 9 grades
// ---------------------------------------------------------------------------
__global__ void pga_prep_weights(const float* __restrict__ Wl,
                                 const float* __restrict__ Wr,
                                 const float* __restrict__ Wout,
                                 __bf16* __restrict__ wpack) {
  int idx = blockIdx.x * 256 + threadIdx.x;       // 3*9*2*4*32*16 = 110592
  if (idx >= 110592) return;
  int t  = idx & 15;
  int ln = (idx >> 4) & 31;
  int nb = (idx >> 9) & 3;
  int kb = (idx >> 11) & 1;
  int wg = idx >> 12;
  int g  = wg % 9;
  int w  = wg / 9;
  const float* W = (w == 0) ? Wl : (w == 1) ? Wr : Wout;
  int o = nb * 16 + (ln & 15);                     // output channel (B-matrix N)
  int i = kb * 32 + (t & 7) + 8 * (ln >> 4) + 16 * (t >> 3); // input channel  (B-matrix K)
  wpack[idx] = (__bf16)W[(o * 64 + i) * 9 + g];
}

// ---------------------------------------------------------------------------
// Fused kernel: equi_linear(l) -> equi_linear(r) -> GP/JOIN bilinear -> equi_linear(out)
// One workgroup = 16 positions; 8 wave32s, 2 blade components per wave.
// ---------------------------------------------------------------------------
__global__ void __launch_bounds__(256)
pga_fused(const float* __restrict__ mv1, const float* __restrict__ mv2,
          const float* __restrict__ bl, const float* __restrict__ br,
          const float* __restrict__ bout, const __bf16* __restrict__ wpack,
          float* __restrict__ out) {
  __shared__ float  bufA[TILE * MROW];   // mv1 -> mv2 -> r -> final D
  __shared__ float  bufB[TILE * MROW];   // l
  __shared__ __bf16 bufO[TILE * MROW];   // bilinear result (bf16, WMMA-A ready)

  const int tid = threadIdx.x;
  const int ln  = tid & 31;
  const int wv  = tid >> 5;
  const int m16 = ln & 15;
  const int hf  = ln >> 4;
  const size_t p0 = (size_t)blockIdx.x * TILE;

  // 1) mv1 tile -> LDS (async DMA, ASYNCcnt)
  async_tile_load(bufA, mv1 + p0 * 1024, tid);
  __syncthreads();

  // 2) l = equi_linear(mv1, Wl, bl) -> bufB
  for (int c = 0; c < 2; ++c) {
    int k = wv * 2 + c;
    v8f acc[4];
    equi_comp(bufA, wpack, 0, bl, k, m16, hf, ln, acc);
#pragma unroll
    for (int nb = 0; nb < 4; ++nb) store_d(bufB, k, nb, m16, hf, acc[nb]);
  }
  __syncthreads();

  // 3) mv2 tile -> LDS (reuse bufA)
  async_tile_load(bufA, mv2 + p0 * 1024, tid);
  __syncthreads();

  // 4) r = equi_linear(mv2, Wr, br), kept in VGPRs until bufA is free
  v8f racc[2][4];
  for (int c = 0; c < 2; ++c) {
    int k = wv * 2 + c;
    equi_comp(bufA, wpack, 1, br, k, m16, hf, ln, racc[c]);
  }
  __syncthreads();

  // 5) r -> bufA
  for (int c = 0; c < 2; ++c) {
    int k = wv * 2 + c;
#pragma unroll
    for (int nb = 0; nb < 4; ++nb) store_d(bufA, k, nb, m16, hf, racc[c][nb]);
  }
  __syncthreads();

  // 6) bilinear: channels 0..31 geometric product, 32..63 join
  for (int q = 0; q < 4; ++q) {
    int pair = tid + q * 256;            // 16 pos x 64 chan
    int m = pair >> 6, cc = pair & 63;
    float lv[16], rv[16], ov[16];
    const float* lp = bufB + m * MROW + cc * 16;
    const float* rp = bufA + m * MROW + cc * 16;
#pragma unroll
    for (int t = 0; t < 16; ++t) { lv[t] = lp[t]; rv[t] = rp[t]; }
    if (cc < 32) bil_eval<true>(lv, rv, ov);
    else         bil_eval<false>(lv, rv, ov);
    __bf16* op = bufO + m * MROW + cc * 16;
#pragma unroll
    for (int t = 0; t < 16; ++t) op[t] = (__bf16)ov[t];
  }
  __syncthreads();

  // 7) final equi_linear(out, Wout, bout) -> bufA
  for (int c = 0; c < 2; ++c) {
    int k = wv * 2 + c;
    v8f acc[4];
    equi_comp(bufO, wpack, 2, bout, k, m16, hf, ln, acc);
#pragma unroll
    for (int nb = 0; nb < 4; ++nb) store_d(bufA, k, nb, m16, hf, acc[nb]);
  }
  __syncthreads();

  // 8) coalesced async store LDS -> global
  async_tile_store(out + p0 * 1024, bufA, tid);
}

// ---------------------------------------------------------------------------
extern "C" void kernel_launch(void* const* d_in, const int* in_sizes, int n_in,
                              void* d_out, int out_size, void* d_ws, size_t ws_size,
                              hipStream_t stream) {
  const float* mv1  = (const float*)d_in[0];
  const float* mv2  = (const float*)d_in[1];
  const float* Wl   = (const float*)d_in[2];
  const float* bl   = (const float*)d_in[3];
  const float* Wr   = (const float*)d_in[4];
  const float* br   = (const float*)d_in[5];
  const float* Wout = (const float*)d_in[6];
  const float* bout = (const float*)d_in[7];
  __bf16* wpack = (__bf16*)d_ws;     // 110592 bf16 = 216 KiB

  pga_prep_weights<<<432, 256, 0, stream>>>(Wl, Wr, Wout, wpack);
  pga_fused<<<1024, 256, 0, stream>>>(mv1, mv2, bl, br, bout, wpack, (float*)d_out);
}
